// ExpertParallelMoELayer_73512660238761
// MI455X (gfx1250) — matrix operand order repaired and checked by reading, not compiled, not run on hardware
//
#include <hip/hip_runtime.h>
#include <hip/hip_bf16.h>
#include <math.h>

#define B_DIM 4
#define S_DIM 4096
#define D_DIM 1024
#define F_DIM 4096
#define E_NUM 8
#define NTOK  (B_DIM * S_DIM)   /* 16384 tokens */

#define MT   32                 /* tokens per block (two 16-row WMMA tiles) */
#define FC   128                /* F chunk per outer iteration              */
#define XSTR 1032               /* 1024 + 8 bf16 pad -> conflict-free b128  */
#define HSTR 136                /* 128 + 8 bf16 pad                         */

#define HS_ELEMS (MT * HSTR)
#define XS_BYTES (MT * XSTR * 2)             /* 66048 */
#define HS_BYTES (HS_ELEMS * 2)              /* 8704  */
#define SMEM_BYTES (XS_BYTES + 2 * HS_BYTES + MT * 4 + MT * 4)

typedef __bf16 bf16_t;
typedef bf16_t v16bf __attribute__((ext_vector_type(16)));
typedef bf16_t v8bf  __attribute__((ext_vector_type(8)));
typedef bf16_t v4bf  __attribute__((ext_vector_type(4)));
typedef float  v8f   __attribute__((ext_vector_type(8)));
typedef float  v4f   __attribute__((ext_vector_type(4)));

__device__ __forceinline__ v16bf cat8(v8bf lo, v8bf hi) {
  return __builtin_shufflevector(lo, hi, 0,1,2,3,4,5,6,7,8,9,10,11,12,13,14,15);
}

__device__ __forceinline__ v16bf lda(const bf16_t* row, int kb, int hi) {
  // contiguous A layout: lanes<16 take K kb..kb+15, lanes>=16 take kb+16..kb+31
  return cat8(*(const v8bf*)(row + kb + hi * 16),
              *(const v8bf*)(row + kb + hi * 16 + 8));
}

// ---------------------------------------------------------------- x -> bf16
__global__ __launch_bounds__(256) void cvt_x_kernel(const float* __restrict__ x,
                                                    bf16_t* __restrict__ xb) {
  int i = (blockIdx.x * 256 + threadIdx.x) * 4;
  v4f v = *(const v4f*)(x + i);
  v4bf o;
  o[0] = (bf16_t)v[0]; o[1] = (bf16_t)v[1]; o[2] = (bf16_t)v[2]; o[3] = (bf16_t)v[3];
  *(v4bf*)(xb + i) = o;
}

// ------------------------------------- [rows][cols] f32 -> [cols][rows] bf16
__global__ __launch_bounds__(256) void transpose_cvt_kernel(const float* __restrict__ src,
                                                            bf16_t* __restrict__ dst,
                                                            int rows, int cols) {
  __shared__ float tile[32][33];
  const size_t mat = (size_t)rows * cols;
  const float* s = src + (size_t)blockIdx.z * mat;
  bf16_t*      d = dst + (size_t)blockIdx.z * mat;
  int c  = blockIdx.x * 32 + threadIdx.x;
  int r0 = blockIdx.y * 32;
  for (int j = threadIdx.y; j < 32; j += 8)
    tile[j][threadIdx.x] = s[(size_t)(r0 + j) * cols + c];
  __syncthreads();
  int nr = blockIdx.x * 32;        // new row = old col
  int nc = r0 + threadIdx.x;       // new col = old row
  for (int j = threadIdx.y; j < 32; j += 8)
    d[(size_t)(nr + j) * rows + nc] = (bf16_t)tile[threadIdx.x][j];
}

// ------------------------------------------------------------------- router
__global__ __launch_bounds__(256) void router_kernel(const float* __restrict__ x,
                                                     const float* __restrict__ Wg,
                                                     int* __restrict__ cnt,
                                                     int* __restrict__ tokL,
                                                     float* __restrict__ gateL) {
  const int lane  = threadIdx.x & 31;
  const int token = blockIdx.x * 8 + (threadIdx.x >> 5);
  const float* xr = x + (size_t)token * D_DIM;
  float acc[E_NUM];
#pragma unroll
  for (int e = 0; e < E_NUM; ++e) acc[e] = 0.f;
  for (int dd = lane; dd < D_DIM; dd += 32) {
    float xv = xr[dd];
    const float* wr = Wg + dd * E_NUM;
#pragma unroll
    for (int e = 0; e < E_NUM; ++e) acc[e] += xv * wr[e];
  }
#pragma unroll
  for (int off = 16; off > 0; off >>= 1) {
#pragma unroll
    for (int e = 0; e < E_NUM; ++e) acc[e] += __shfl_xor(acc[e], off, 32);
  }
  if (lane == 0) {
    int e0 = 0; float v0 = acc[0];
#pragma unroll
    for (int e = 1; e < E_NUM; ++e) if (acc[e] > v0) { v0 = acc[e]; e0 = e; }
    int e1 = -1; float v1 = -3.4e38f;
#pragma unroll
    for (int e = 0; e < E_NUM; ++e) if (e != e0 && acc[e] > v1) { v1 = acc[e]; e1 = e; }
    float g0 = 1.f / (1.f + __expf(v1 - v0));   // renormalized top-2 gates
    float g1 = 1.f - g0;
    int s0 = atomicAdd(&cnt[e0], 1);
    tokL[e0 * NTOK + s0] = token; gateL[e0 * NTOK + s0] = g0;
    int s1 = atomicAdd(&cnt[e1], 1);
    tokL[e1 * NTOK + s1] = token; gateL[e1 * NTOK + s1] = g1;
  }
}

// ------------------------------------------------------------- expert FFN
__global__ __launch_bounds__(256) void moe_ffn_kernel(
    const bf16_t* __restrict__ xb,     // [NTOK][D]
    const bf16_t* __restrict__ W1b,    // [E][F][D]  col-major per F column
    const bf16_t* __restrict__ W2b,    // [E][D][F]  col-major per D column
    const float*  __restrict__ b1,     // [E][F]
    const float*  __restrict__ b2,     // [E][D]
    const int*    __restrict__ cnt,
    const int*    __restrict__ tokL,
    const float*  __restrict__ gateL,
    float*        __restrict__ out) {
  extern __shared__ __align__(16) char smem[];
  bf16_t* Xs      = (bf16_t*)smem;                              // [MT][XSTR]
  bf16_t* Hs      = (bf16_t*)(smem + XS_BYTES);                 // 2 x [MT][HSTR]
  int*    toks_s  = (int*)  (smem + XS_BYTES + 2 * HS_BYTES);   // [MT]
  float*  gates_s = (float*)(smem + XS_BYTES + 2 * HS_BYTES + MT * 4);

  const int e     = blockIdx.y;
  const int tile  = blockIdx.x;
  const int count = cnt[e];
  if (tile * MT >= count) return;

  const int tid  = threadIdx.x;
  const int wave = tid >> 5;
  const int lane = tid & 31;
  const int ln   = lane & 15;
  const int hi   = lane >> 4;

  if (tid < MT) {
    int idx = tile * MT + tid;
    int t; float g;
    if (idx < count) { t = tokL[e * NTOK + idx]; g = gateL[e * NTOK + idx]; }
    else             { t = tokL[e * NTOK];       g = 0.f; }
    toks_s[tid] = t; gates_s[tid] = g;
  }
  __syncthreads();

  // stage X tile (MT x 1024 bf16) into LDS
  for (int i = tid; i < MT * (D_DIM / 8); i += 256) {
    int r = i >> 7;
    int c = (i & 127) << 3;
    *(v8bf*)&Xs[r * XSTR + c] = *(const v8bf*)(xb + (size_t)toks_s[r] * D_DIM + c);
  }
  __syncthreads();

  const v8f vzero = {0.f,0.f,0.f,0.f,0.f,0.f,0.f,0.f};
  v8f acc0[8], acc1[8];                 // rows 0-15 / rows 16-31
#pragma unroll
  for (int t = 0; t < 8; ++t) { acc0[t] = vzero; acc1[t] = vzero; }

  const bf16_t* xrow0 = &Xs[ln * XSTR];
  const bf16_t* xrow1 = &Xs[(16 + ln) * XSTR];
  const int     nbase = wave * 128;     // this wave's 128 output cols

  int ibuf = 0;
  for (int fo = 0; fo < F_DIM; fo += FC, ibuf ^= 1) {
    bf16_t* Hb = Hs + ibuf * HS_ELEMS;  // ping-pong H buffer

    // ---- GEMM1: H[:, fo + wave*16 .. +15], both M halves share B ----
    v8f h0 = vzero, h1 = vzero;
    const bf16_t* w1p =
        W1b + ((size_t)e * F_DIM + fo + wave * 16 + ln) * D_DIM + hi * 16;
#pragma unroll 2
    for (int kb = 0; kb < D_DIM; kb += 32) {
      v16bf b  = *(const v16bf*)(w1p + kb);
      v16bf a0 = lda(xrow0, kb, hi);
      v16bf a1 = lda(xrow1, kb, hi);
      h0 = __builtin_amdgcn_wmma_f32_16x16x32_bf16(false, a0, false, b,
                                                   (short)0, h0, false, false);
      h1 = __builtin_amdgcn_wmma_f32_16x16x32_bf16(false, a1, false, b,
                                                   (short)0, h1, false, false);
    }
    // bias + ReLU, write bf16 H tile to LDS
    {
      const int   hn  = wave * 16 + ln;
      const float b1v = b1[e * F_DIM + fo + hn];
#pragma unroll
      for (int r = 0; r < 8; ++r) {
        float v0 = h0[r] + b1v; v0 = v0 > 0.f ? v0 : 0.f;
        float v1 = h1[r] + b1v; v1 = v1 > 0.f ? v1 : 0.f;
        Hb[(hi * 8 + r) * HSTR + hn]      = (bf16_t)v0;  // M = r+hi*8
        Hb[(16 + hi * 8 + r) * HSTR + hn] = (bf16_t)v1;  // M+16
      }
    }
    __syncthreads();   // single barrier per chunk (double-buffered Hs)

    // ---- GEMM2: acc += H(MTxFC) @ W2[fo:fo+FC, nbase:nbase+128) ----
    const bf16_t* hrow0 = Hb + ln * HSTR;
    const bf16_t* hrow1 = Hb + (16 + ln) * HSTR;
#pragma unroll
    for (int kb = 0; kb < FC; kb += 32) {
      v16bf a0 = lda(hrow0, kb, hi);
      v16bf a1 = lda(hrow1, kb, hi);
#pragma unroll
      for (int t = 0; t < 8; ++t) {
        const bf16_t* w2p =
            W2b + ((size_t)e * D_DIM + nbase + t * 16 + ln) * F_DIM
                + fo + kb + hi * 16;
        v16bf b = *(const v16bf*)w2p;   // one B fragment feeds two WMMAs
        acc0[t] = __builtin_amdgcn_wmma_f32_16x16x32_bf16(false, a0, false, b,
                                                          (short)0, acc0[t],
                                                          false, false);
        acc1[t] = __builtin_amdgcn_wmma_f32_16x16x32_bf16(false, a1, false, b,
                                                          (short)0, acc1[t],
                                                          false, false);
      }
    }
    // no trailing barrier: next chunk writes the other H buffer
  }

  // ---- epilogue: +b2, *gate, scatter-add ----
#pragma unroll
  for (int t = 0; t < 8; ++t) {
    const int   n   = nbase + t * 16 + ln;
    const float b2v = b2[e * D_DIM + n];
#pragma unroll
    for (int r = 0; r < 8; ++r) {
      const int m0 = hi * 8 + r;
      const int m1 = 16 + m0;
      atomicAdd(out + (size_t)toks_s[m0] * D_DIM + n, (acc0[t][r] + b2v) * gates_s[m0]);
      atomicAdd(out + (size_t)toks_s[m1] * D_DIM + n, (acc1[t][r] + b2v) * gates_s[m1]);
    }
  }
}

// ---------------------------------------------------------------- launcher
extern "C" void kernel_launch(void* const* d_in, const int* in_sizes, int n_in,
                              void* d_out, int out_size, void* d_ws, size_t ws_size,
                              hipStream_t stream) {
  (void)in_sizes; (void)n_in;
  const float* x  = (const float*)d_in[0];
  const float* Wg = (const float*)d_in[1];
  const float* W1 = (const float*)d_in[2];
  const float* b1 = (const float*)d_in[3];
  const float* W2 = (const float*)d_in[4];
  const float* b2 = (const float*)d_in[5];
  float* out = (float*)d_out;

  char* ws = (char*)d_ws;
  const size_t SZ_W1B = (size_t)E_NUM * F_DIM * D_DIM * sizeof(bf16_t); // 64 MiB
  const size_t SZ_W2B = SZ_W1B;                                         // 64 MiB
  const size_t SZ_XB  = (size_t)NTOK * D_DIM * sizeof(bf16_t);          // 32 MiB
  const size_t SZ_TOK = (size_t)E_NUM * NTOK * sizeof(int);
  const size_t SZ_GT  = (size_t)E_NUM * NTOK * sizeof(float);
  const size_t OFF_W1B = 0;
  const size_t OFF_W2B = OFF_W1B + SZ_W1B;
  const size_t OFF_XB  = OFF_W2B + SZ_W2B;
  const size_t OFF_TOK = OFF_XB  + SZ_XB;
  const size_t OFF_GT  = OFF_TOK + SZ_TOK;
  const size_t OFF_CNT = OFF_GT  + SZ_GT;
  if (ws_size < OFF_CNT + 256) return;   // workspace too small: bail out

  bf16_t* W1b  = (bf16_t*)(ws + OFF_W1B);
  bf16_t* W2b  = (bf16_t*)(ws + OFF_W2B);
  bf16_t* xb   = (bf16_t*)(ws + OFF_XB);
  int*    tokL = (int*)   (ws + OFF_TOK);
  float*  gateL= (float*) (ws + OFF_GT);
  int*    cnt  = (int*)   (ws + OFF_CNT);

  hipMemsetAsync(out, 0, (size_t)out_size * sizeof(float), stream);
  hipMemsetAsync(cnt, 0, E_NUM * sizeof(int), stream);

  cvt_x_kernel<<<(NTOK * D_DIM) / (256 * 4), 256, 0, stream>>>(x, xb);

  dim3 tb(32, 8);
  transpose_cvt_kernel<<<dim3(F_DIM / 32, D_DIM / 32, E_NUM), tb, 0, stream>>>(
      W1, W1b, D_DIM, F_DIM);   // [D,F] -> [F,D]
  transpose_cvt_kernel<<<dim3(D_DIM / 32, F_DIM / 32, E_NUM), tb, 0, stream>>>(
      W2, W2b, F_DIM, D_DIM);   // [F,D] -> [D,F]

  router_kernel<<<NTOK / 8, 256, 0, stream>>>(x, Wg, cnt, tokL, gateL);

  moe_ffn_kernel<<<dim3(NTOK / MT, E_NUM), 256, SMEM_BYTES, stream>>>(
      xb, W1b, W2b, b1, b2, cnt, tokL, gateL, out);
}
